// MultiHeadAttn_55525337203105
// MI455X (gfx1250) — matrix operand order repaired and checked by reading, not compiled, not run on hardware
//
#include <hip/hip_runtime.h>

// ---------------------------------------------------------------------------
// MI455X (gfx1250) multi-head attention: bf16 WMMA GEMMs + flash attention.
// ---------------------------------------------------------------------------

typedef __attribute__((ext_vector_type(16))) __bf16          v16bf;
typedef __attribute__((ext_vector_type(16))) unsigned short  v16u;
typedef __attribute__((ext_vector_type(8)))  unsigned short  v8u;
typedef __attribute__((ext_vector_type(4)))  unsigned short  v4u;
typedef __attribute__((ext_vector_type(8)))  float           v8f;

#define B_     2
#define S_     2048
#define D_     1024
#define H_     16
#define DEPTH_ 64
#define M_     (B_ * S_)   // 4096 rows

// Async global->LDS path (GLOBAL_LOAD_ASYNC_TO_LDS_B128, ASYNCcnt) if the
// toolchain exposes it; otherwise fall back to load+ds_store.
#if __has_builtin(__builtin_amdgcn_global_load_async_to_lds_b128)
#define HAVE_ASYNC_LDS 1
// builtin signature (from hipcc diagnostics): (v4i addrspace(1)*, v4i addrspace(3)*, imm, imm)
typedef int v4i_vs __attribute__((vector_size(16)));
typedef __attribute__((address_space(1))) v4i_vs* gas1_v4i;
typedef __attribute__((address_space(3))) v4i_vs* las3_v4i;
#else
#define HAVE_ASYNC_LDS 0
#endif

__device__ __forceinline__ unsigned short f32_to_bf16(float f) {
  unsigned int u = __builtin_bit_cast(unsigned int, f);
  u += 0x7FFFu + ((u >> 16) & 1u);      // round-to-nearest-even
  return (unsigned short)(u >> 16);
}

__device__ __forceinline__ v16bf mkfrag(v8u lo, v8u hi) {
  v16u t = __builtin_shufflevector(lo, hi, 0,1,2,3,4,5,6,7,8,9,10,11,12,13,14,15);
  return __builtin_bit_cast(v16bf, t);
}

// reductions across a 16-lane half (xor masks < 16 never cross halves)
__device__ __forceinline__ float hred_max(float v) {
  v = fmaxf(v, __shfl_xor(v, 1, 32));
  v = fmaxf(v, __shfl_xor(v, 2, 32));
  v = fmaxf(v, __shfl_xor(v, 4, 32));
  v = fmaxf(v, __shfl_xor(v, 8, 32));
  return v;
}
__device__ __forceinline__ float hred_sum(float v) {
  v += __shfl_xor(v, 1, 32);
  v += __shfl_xor(v, 2, 32);
  v += __shfl_xor(v, 4, 32);
  v += __shfl_xor(v, 8, 32);
  return v;
}

// ---------------------------------------------------------------------------
// GEMM: C[m][n] = sum_k X[m][k] * W[n][k] + bias[n]   (X: [4096,1024] f32,
// W: [1024,1024] f32).  Both operands are traversed along contiguous k, so
// both map naturally onto WMMA A/B fragment layouts after bf16 conversion.
// PROJ=true : store bf16 into head-split [B,H,S,64] workspace
// PROJ=false: store f32 row-major into d_out
// ---------------------------------------------------------------------------
template <bool PROJ>
__launch_bounds__(256)
__global__ void gemm_kernel(const float* __restrict__ X, const float* __restrict__ W,
                            const float* __restrict__ bias,
                            unsigned short* __restrict__ outBf,
                            float* __restrict__ outF) {
  __shared__ unsigned short As[128][40];   // 128 rows x 32 k (bf16), padded
  __shared__ unsigned short Bs[128][40];

  const int tid  = threadIdx.x;
  const int wid  = tid >> 5, lane = tid & 31;
  const int l16  = lane & 15, hf = lane >> 4;
  const int waveM = wid >> 1;              // 0..3 -> 32-row strip
  const int waveN = wid & 1;               // 0..1 -> 64-col strip
  const int mBase = blockIdx.y * 128;
  const int nBase = blockIdx.x * 128;

  const int ldRow = tid >> 1;              // 0..127
  const int ldK   = (tid & 1) * 16;        // 0 or 16

  const v8f vzero = {0.f,0.f,0.f,0.f,0.f,0.f,0.f,0.f};
  v8f acc[2][4];
  #pragma unroll
  for (int mi = 0; mi < 2; ++mi)
    #pragma unroll
    for (int ni = 0; ni < 4; ++ni) acc[mi][ni] = vzero;

  for (int k0 = 0; k0 < D_; k0 += 32) {
    // cooperative tile load with on-the-fly f32 -> bf16 conversion
    const float4* xa = (const float4*)(X + (size_t)(mBase + ldRow) * D_ + k0 + ldK);
    const float4* wb = (const float4*)(W + (size_t)(nBase + ldRow) * D_ + k0 + ldK);
    #pragma unroll
    for (int q = 0; q < 4; ++q) {
      float4 xv = xa[q];
      float4 wv = wb[q];
      v4u xu = { f32_to_bf16(xv.x), f32_to_bf16(xv.y), f32_to_bf16(xv.z), f32_to_bf16(xv.w) };
      v4u wu = { f32_to_bf16(wv.x), f32_to_bf16(wv.y), f32_to_bf16(wv.z), f32_to_bf16(wv.w) };
      *(v4u*)&As[ldRow][ldK + 4*q] = xu;
      *(v4u*)&Bs[ldRow][ldK + 4*q] = wu;
    }
    __syncthreads();

    // A fragment: lane l holds row (l%16), k = i + (i&8?8:0) + (l/16)*8
    v16bf afrag[2], bfrag[4];
    #pragma unroll
    for (int mi = 0; mi < 2; ++mi) {
      const int row = waveM*32 + mi*16 + l16;
      v8u lo = *(const v8u*)&As[row][hf*8];
      v8u hi = *(const v8u*)&As[row][16 + hf*8];
      afrag[mi] = mkfrag(lo, hi);
    }
    // B fragment: lane l holds col (l%16), k = i + (i&8?8:0) + (l/16)*8
    #pragma unroll
    for (int ni = 0; ni < 4; ++ni) {
      const int col = waveN*64 + ni*16 + l16;
      v8u lo = *(const v8u*)&Bs[col][hf*8];
      v8u hi = *(const v8u*)&Bs[col][16 + hf*8];
      bfrag[ni] = mkfrag(lo, hi);
    }
    #pragma unroll
    for (int mi = 0; mi < 2; ++mi)
      #pragma unroll
      for (int ni = 0; ni < 4; ++ni)
        acc[mi][ni] = __builtin_amdgcn_wmma_f32_16x16x32_bf16(
            false, afrag[mi], false, bfrag[ni], (short)0, acc[mi][ni], false, false);
    __syncthreads();
  }

  // store: C layout -> VGPR r, lanes 0-15 row r, lanes 16-31 row r+8
  #pragma unroll
  for (int mi = 0; mi < 2; ++mi) {
    #pragma unroll
    for (int ni = 0; ni < 4; ++ni) {
      const int nGlob = nBase + waveN*64 + ni*16 + l16;
      const float bv = bias[nGlob];
      #pragma unroll
      for (int r = 0; r < 8; ++r) {
        const int   mGlob = mBase + waveM*32 + mi*16 + r + 8*hf;
        const float val   = acc[mi][ni][r] + bv;
        if (PROJ) {
          const int b = mGlob >> 11, s = mGlob & 2047;
          const int h = nGlob >> 6,  d = nGlob & 63;
          outBf[(((size_t)b*H_ + h)*S_ + s)*DEPTH_ + d] = f32_to_bf16(val);
        } else {
          outF[(size_t)mGlob * D_ + nGlob] = val;
        }
      }
    }
  }
}

// ---------------------------------------------------------------------------
// Flash attention: block = 4 waves (128 threads), each wave owns 16 query
// rows of one (b,h); streams 32-key K/V tiles; online softmax; mask read once.
// K tile goes global->LDS via the async data path (ASYNCcnt) when available.
// ---------------------------------------------------------------------------
__launch_bounds__(128)
__global__ void attn_kernel(const unsigned short* __restrict__ Qh,
                            const unsigned short* __restrict__ Kh,
                            const unsigned short* __restrict__ Vh,
                            const float* __restrict__ mask,
                            float* __restrict__ attnOut) {
  __shared__ unsigned short Ks[32][72];      // [key][depth] bf16
  __shared__ unsigned short Vt[64][40];      // [depth][key] bf16 (transposed)
  __shared__ unsigned short Ps[4][16][40];   // per-wave P tile [qrow][key]

  const int tid  = threadIdx.x;
  const int wid  = tid >> 5, lane = tid & 31;
  const int l16  = lane & 15, hf = lane >> 4;
  const int bh   = blockIdx.y;                       // b*H + h
  const int qRow0 = blockIdx.x * 64 + wid * 16;

  const size_t headOff = (size_t)bh * S_ * DEPTH_;
  const unsigned short* Qb = Qh + headOff;
  const unsigned short* Kb = Kh + headOff;
  const unsigned short* Vb = Vh + headOff;
  const float* maskB = mask + (size_t)bh * S_ * S_;

  // Q: 16 rows x 64 depth -> two A fragments (depth 0..31, 32..63)
  v16bf qf[2];
  {
    const unsigned short* qrow = Qb + (size_t)(qRow0 + l16) * DEPTH_;
    #pragma unroll
    for (int kc = 0; kc < 2; ++kc) {
      v8u lo = *(const v8u*)(qrow + kc*32 + hf*8);
      v8u hi = *(const v8u*)(qrow + kc*32 + 16 + hf*8);
      qf[kc] = mkfrag(lo, hi);
    }
  }

  const v8f vzero = {0.f,0.f,0.f,0.f,0.f,0.f,0.f,0.f};
  v8f O[4];
  #pragma unroll
  for (int j = 0; j < 4; ++j) O[j] = vzero;
  float mrow[8], lrow[8];
  #pragma unroll
  for (int r = 0; r < 8; ++r) { mrow[r] = -3.0e38f; lrow[r] = 0.f; }

  const int ldRow = tid >> 2;          // key row 0..31
  const int ldSeg = (tid & 3) * 16;    // depth segment

  for (int kt = 0; kt < S_/32; ++kt) {
    const int keyBase = kt * 32;
    // ---- cooperative K/V tile load (V transposed in LDS) ----
    {
      const unsigned short* ksrc = Kb + (size_t)(keyBase + ldRow)*DEPTH_ + ldSeg;
#if HAVE_ASYNC_LDS
      // direct global->LDS DMA, bypasses VGPRs, tracked by ASYNCcnt
      __builtin_amdgcn_global_load_async_to_lds_b128(
          (gas1_v4i)(void*)ksrc,
          (las3_v4i)(void*)&Ks[ldRow][ldSeg], 0, 0);
      __builtin_amdgcn_global_load_async_to_lds_b128(
          (gas1_v4i)(void*)(ksrc + 8),
          (las3_v4i)(void*)&Ks[ldRow][ldSeg + 8], 0, 0);
#else
      v8u klo = *(const v8u*)ksrc;
      v8u khi = *(const v8u*)(ksrc + 8);
      *(v8u*)&Ks[ldRow][ldSeg]     = klo;
      *(v8u*)&Ks[ldRow][ldSeg + 8] = khi;
#endif
      const unsigned short* vsrc = Vb + (size_t)(keyBase + ldRow)*DEPTH_ + ldSeg;
      v8u vlo = *(const v8u*)vsrc;
      v8u vhi = *(const v8u*)(vsrc + 8);
      #pragma unroll
      for (int i = 0; i < 8; ++i) Vt[ldSeg + i][ldRow]     = vlo[i];
      #pragma unroll
      for (int i = 0; i < 8; ++i) Vt[ldSeg + 8 + i][ldRow] = vhi[i];
      if (kt + 1 < S_/32) {
        __builtin_prefetch(ksrc + (size_t)32*DEPTH_, 0, 1);   // global_prefetch_b8
        __builtin_prefetch(vsrc + (size_t)32*DEPTH_, 0, 1);
      }
    }
#if HAVE_ASYNC_LDS
#if __has_builtin(__builtin_amdgcn_s_wait_asynccnt)
    __builtin_amdgcn_s_wait_asynccnt(0);
#else
    asm volatile("s_wait_asynccnt 0x0" ::: "memory");
#endif
#endif
    __syncthreads();

    // ---- scores: S = Q K^T, two 16x16 C frags (keys 0-15, 16-31) ----
    v8f s[2];
    #pragma unroll
    for (int nb = 0; nb < 2; ++nb) {
      s[nb] = vzero;
      #pragma unroll
      for (int kc = 0; kc < 2; ++kc) {
        const int kl = nb*16 + l16;
        v8u lo = *(const v8u*)&Ks[kl][kc*32 + hf*8];
        v8u hi = *(const v8u*)&Ks[kl][kc*32 + 16 + hf*8];
        v16bf kf = mkfrag(lo, hi);
        s[nb] = __builtin_amdgcn_wmma_f32_16x16x32_bf16(
            false, qf[kc], false, kf, (short)0, s[nb], false, false);
      }
    }

    // ---- scale + mask + online softmax (per-row stats replicated per half) --
    #pragma unroll
    for (int r = 0; r < 8; ++r) {
      const int    qm      = qRow0 + r + 8*hf;
      const size_t mOff    = (size_t)qm * S_ + keyBase;
      const float  m0      = maskB[mOff + l16];
      const float  m1      = maskB[mOff + 16 + l16];
      const float  s0      = s[0][r]*0.125f + m0*(-1e9f);
      const float  s1      = s[1][r]*0.125f + m1*(-1e9f);
      const float  tm      = hred_max(fmaxf(s0, s1));
      const float  mnew    = fmaxf(mrow[r], tm);
      const float  alpha   = __expf(mrow[r] - mnew);
      mrow[r] = mnew;
      const float  p0      = __expf(s0 - mnew);
      const float  p1      = __expf(s1 - mnew);
      lrow[r] = lrow[r]*alpha + hred_sum(p0 + p1);
      #pragma unroll
      for (int j = 0; j < 4; ++j) O[j][r] *= alpha;
      const int pm = r + 8*hf;
      Ps[wid][pm][l16]      = f32_to_bf16(p0);
      Ps[wid][pm][16 + l16] = f32_to_bf16(p1);
    }
    asm volatile("s_wait_dscnt 0x0" ::: "memory");  // wave-local P visible

    // ---- O += P V  (P: 16x32 A frag from LDS, V: 32x16 B frags, 4 WMMA) ----
    v16bf pf;
    {
      v8u lo = *(const v8u*)&Ps[wid][l16][hf*8];
      v8u hi = *(const v8u*)&Ps[wid][l16][16 + hf*8];
      pf = mkfrag(lo, hi);
    }
    #pragma unroll
    for (int j = 0; j < 4; ++j) {
      const int dcol = j*16 + l16;
      v8u lo = *(const v8u*)&Vt[dcol][hf*8];
      v8u hi = *(const v8u*)&Vt[dcol][16 + hf*8];
      v16bf vf = mkfrag(lo, hi);
      O[j] = __builtin_amdgcn_wmma_f32_16x16x32_bf16(
          false, pf, false, vf, (short)0, O[j], false, false);
    }
    __syncthreads();
  }

  // ---- normalize + write attn_products f32 [B,S,H,64] ----
  const int b = bh >> 4, h = bh & 15;
  #pragma unroll
  for (int r = 0; r < 8; ++r) {
    const float inv  = 1.0f / lrow[r];
    const int   srow = qRow0 + r + 8*hf;
    float* orow = attnOut + (((size_t)b*S_ + srow)*H_ + h)*DEPTH_;
    #pragma unroll
    for (int j = 0; j < 4; ++j) orow[j*16 + l16] = O[j][r] * inv;
  }
}

// ---------------------------------------------------------------------------
extern "C" void kernel_launch(void* const* d_in, const int* in_sizes, int n_in,
                              void* d_out, int out_size, void* d_ws, size_t ws_size,
                              hipStream_t stream) {
  (void)in_sizes; (void)n_in; (void)out_size; (void)ws_size;
  const float* q    = (const float*)d_in[0];
  const float* k    = (const float*)d_in[1];
  const float* v    = (const float*)d_in[2];
  const float* mask = (const float*)d_in[3];
  const float* wq_w = (const float*)d_in[4];
  const float* wq_b = (const float*)d_in[5];
  const float* wk_w = (const float*)d_in[6];
  const float* wk_b = (const float*)d_in[7];
  const float* wv_w = (const float*)d_in[8];
  const float* wv_b = (const float*)d_in[9];
  const float* fc_w = (const float*)d_in[10];
  const float* fc_b = (const float*)d_in[11];

  unsigned short* qh = (unsigned short*)d_ws;              // [B,H,S,64] bf16
  unsigned short* kh = qh + (size_t)M_ * D_;
  unsigned short* vh = kh + (size_t)M_ * D_;

  float* out  = (float*)d_out;                             // outputs  [B,S,D]
  float* attn = out + (size_t)M_ * D_;                     // attn_products

  dim3 gGrid(D_/128, M_/128);   // (8, 32)
  gemm_kernel<true ><<<gGrid, 256, 0, stream>>>(q, wq_w, wq_b, qh, nullptr);
  gemm_kernel<true ><<<gGrid, 256, 0, stream>>>(k, wk_w, wk_b, kh, nullptr);
  gemm_kernel<true ><<<gGrid, 256, 0, stream>>>(v, wv_w, wv_b, vh, nullptr);

  dim3 aGrid(S_/64, B_*H_);     // (32, 32)
  attn_kernel<<<aGrid, 128, 0, stream>>>(qh, kh, vh, mask, attn);

  gemm_kernel<false><<<gGrid, 256, 0, stream>>>(attn, fc_w, fc_b, nullptr, out);
}
